// SpatialTemporalBlock_5257039970450
// MI455X (gfx1250) — matrix operand order repaired and checked by reading, not compiled, not run on hardware
//
#include <hip/hip_runtime.h>
#include <cstddef>
#include <cstdint>

// ---------------------------------------------------------------------------
// SpatialTemporalBlock for MI455X (gfx1250, wave32, WMMA).
// All 1x1 convolutions run as bf16 WMMA GEMMs (v_wmma_f32_16x16x32_bf16),
// activations kept in bf16 (everything stays L2-resident on a 192MB L2).
// One wave computes a 16(M) x 128(N) output tile: 8 WMMAs per 32-wide K-step.
// ---------------------------------------------------------------------------

typedef __bf16 bf16;
typedef __attribute__((ext_vector_type(16))) __bf16 v16bf;
typedef __attribute__((ext_vector_type(8)))  __bf16 v8bf;
typedef __attribute__((ext_vector_type(8)))  float  v8f;

static constexpr int BB = 32, CIN = 128, V = 25, T = 128, VT = V * T;   // 3200
static constexpr int DIM1 = 256, BC = 42;
static constexpr long NTOT = (long)BB * VT;                              // 102400

static __device__ __forceinline__ float bf2f(bf16 h) {
    unsigned short s = __builtin_bit_cast(unsigned short, h);
    unsigned u = ((unsigned)s) << 16;
    return __builtin_bit_cast(float, u);
}
static __device__ __forceinline__ bf16 f2bf(float f) {
    unsigned u = __builtin_bit_cast(unsigned, f);
    u += 0x7fffu + ((u >> 16) & 1u);                 // round-to-nearest-even
    unsigned short s = (unsigned short)(u >> 16);
    return __builtin_bit_cast(bf16, s);
}
static __device__ __forceinline__ void stv(float* p, float v) { *p = v; }
static __device__ __forceinline__ void stv(bf16* p, float v) { *p = f2bf(v); }

// ---------------------------------------------------------------------------
// bf16 WMMA GEMM:  out[b,o,n] = sum_k Wp[o,k] * X[b,k,n]  (+bias[o]) (+add)
//   Wp : padded bf16 weights, Opad x Kpad row-major (Kpad mult of 32)
//   X  : bf16 activations (B, Kpad, VT)
//   one wave computes a 16(M) x 128(N) tile; grid = (NTOT/128, Opad/16)
// ---------------------------------------------------------------------------
template <typename OutT, bool HB, bool HA>
__global__ __launch_bounds__(32) void k_gemm(const bf16* __restrict__ Wp,
                                             const bf16* __restrict__ X,
                                             const float* __restrict__ bias,
                                             const bf16* __restrict__ add,
                                             OutT* __restrict__ out,
                                             int Kpad, int O) {
    const int lane = threadIdx.x;
    const int b    = blockIdx.x / 25;          // 3200/128 = 25 tiles per batch
    const int n0   = (blockIdx.x % 25) * 128;
    const int o0   = blockIdx.y * 16;

    v8f acc[8];
#pragma unroll
    for (int j = 0; j < 8; ++j) acc[j] = 0.f;

    // A fragment: lane L -> row M=L&15; K elems 16*(e/8) + 8*(L>>4) + e%8
    const bf16* wrow = Wp + (size_t)(o0 + (lane & 15)) * Kpad + (lane >> 4) * 8;
    // B fragment: lane L -> N col L&15; K elems 16*(e/8) + 8*(L>>4) + e%8
    const bf16* xrow = X + ((size_t)b * Kpad + (lane >> 4) * 8) * VT + n0 + (lane & 15);

    for (int c0 = 0; c0 < Kpad; c0 += 32) {
        v8bf wlo = *(const v8bf*)(wrow + c0);
        v8bf whi = *(const v8bf*)(wrow + c0 + 16);
        v16bf af;
#pragma unroll
        for (int i = 0; i < 8; ++i) { af[i] = wlo[i]; af[i + 8] = whi[i]; }

        const bf16* xc = xrow + (size_t)c0 * VT;
#pragma unroll
        for (int j = 0; j < 8; ++j) {
            v16bf bf;
#pragma unroll
            for (int i = 0; i < 8; ++i) { bf[i] = xc[(size_t)i * VT + j * 16]; bf[i + 8] = xc[(size_t)(16 + i) * VT + j * 16]; }
            acc[j] = __builtin_amdgcn_wmma_f32_16x16x32_bf16(false, af, false, bf,
                                                             (short)0, acc[j], false, false);
        }
    }

    // C/D layout: VGPR r, lanes 0-15 -> M=r, N=lane; lanes 16-31 -> M=8+r
    const int ncol  = lane & 15;
    const int rbase = (lane >> 4) * 8;
#pragma unroll
    for (int r = 0; r < 8; ++r) {
        int o = o0 + rbase + r;
        if (o < O) {
            float bi = HB ? bias[o] : 0.f;
            size_t ri = ((size_t)b * O + o) * VT + n0 + ncol;
#pragma unroll
            for (int j = 0; j < 8; ++j) {
                float v = acc[j][r] + bi;
                if (HA) v += bf2f(add[ri + j * 16]);
                stv(out + ri + j * 16, v);
            }
        }
    }
}

// ---------------------------------------------------------------------------
// Utility / elementwise kernels
// ---------------------------------------------------------------------------
__global__ void k_f2b(const float* __restrict__ in, bf16* __restrict__ out, long n) {
    long i = (long)blockIdx.x * blockDim.x + threadIdx.x;
    if (i < n) out[i] = f2bf(in[i]);
}

__global__ void k_cvt_w(const float* __restrict__ W, bf16* __restrict__ Wp,
                        int O, int K, int Opad, int Kpad) {
    int i = blockIdx.x * blockDim.x + threadIdx.x;
    int tot = Opad * Kpad;
    if (i >= tot) return;
    int o = i / Kpad, k = i % Kpad;
    float v = (o < O && k < K) ? W[o * K + k] : 0.f;
    Wp[i] = f2bf(v);
}

__global__ void k_zeropad(bf16* __restrict__ concat) {   // zero channels 252..255
    long idx = (long)blockIdx.x * blockDim.x + threadIdx.x;
    long tot = (long)BB * 4 * VT;
    if (idx >= tot) return;
    int n = (int)(idx % VT); long r = idx / VT;
    int c = (int)(r % 4);    int b = (int)(r / 4);
    concat[((long)b * 256 + 252 + c) * VT + n] = f2bf(0.f);
}

// training-mode BN stats per channel (reduce over B * L elements)
__global__ __launch_bounds__(256) void k_bn_stats(const bf16* __restrict__ x,
                                                  int CH, int L,
                                                  float* __restrict__ mean,
                                                  float* __restrict__ rsig) {
    __shared__ float ss[256], sq[256];
    int ch = blockIdx.x;
    long cnt = (long)BB * L;
    float s = 0.f, q = 0.f;
    for (long i = threadIdx.x; i < cnt; i += blockDim.x) {
        long b = i / L, n = i % L;
        float v = bf2f(x[(b * CH + ch) * L + n]);
        s += v; q += v * v;
    }
    ss[threadIdx.x] = s; sq[threadIdx.x] = q;
    __syncthreads();
    for (int st = 128; st > 0; st >>= 1) {
        if ((int)threadIdx.x < st) {
            ss[threadIdx.x] += ss[threadIdx.x + st];
            sq[threadIdx.x] += sq[threadIdx.x + st];
        }
        __syncthreads();
    }
    if (threadIdx.x == 0) {
        float m = ss[0] / (float)cnt;
        float var = sq[0] / (float)cnt - m * m;
        mean[ch] = m;
        rsig[ch] = rsqrtf(var + 1e-5f);
    }
}

// y = [relu]( (x-mean)*rsig*gamma + beta )  [+ add];  out may have channel offset
template <bool RELU, bool HA>
__global__ void k_bn_apply(const bf16* __restrict__ x, int CH, int L,
                           const float* __restrict__ mean, const float* __restrict__ rsig,
                           const float* __restrict__ gamma, const float* __restrict__ beta,
                           const bf16* __restrict__ add, bf16* __restrict__ out,
                           int OCT, int OCO) {
    long idx = (long)blockIdx.x * blockDim.x + threadIdx.x;
    long tot = (long)BB * CH * L;
    if (idx >= tot) return;
    int n = (int)(idx % L); long r = idx / L;
    int ch = (int)(r % CH); int b = (int)(r / CH);
    float v = (bf2f(x[idx]) - mean[ch]) * rsig[ch] * gamma[ch] + beta[ch];
    if (RELU) v = fmaxf(v, 0.f);
    long oidx = ((long)b * OCT + OCO + ch) * L + n;
    if (HA) v += bf2f(add[oidx]);
    out[oidx] = f2bf(v);
}

// g[b,t,v,w] = softmax_w( sum_c g1[b,c,v,t] * g2[b,c,w,t] )  -- LDS staged
__global__ __launch_bounds__(128) void k_gmat(const bf16* __restrict__ g1,
                                              const bf16* __restrict__ g2,
                                              float* __restrict__ gout) {
    __shared__ float s1[DIM1 * V];
    __shared__ float s2[DIM1 * V];
    __shared__ float S[V * V];
    int bt = blockIdx.x;
    int b = bt / T, t = bt % T;
    for (int i = threadIdx.x; i < DIM1 * V; i += blockDim.x) {
        int c = i / V, v = i % V;
        long src = (((long)b * DIM1 + c) * V + v) * T + t;
        s1[v * DIM1 + c] = bf2f(g1[src]);
        s2[v * DIM1 + c] = bf2f(g2[src]);
    }
    __syncthreads();
    for (int e = threadIdx.x; e < V * V; e += blockDim.x) {
        int v = e / V, w = e % V;
        float acc = 0.f;
        for (int c = 0; c < DIM1; ++c) acc += s1[v * DIM1 + c] * s2[w * DIM1 + c];
        S[e] = acc;
    }
    __syncthreads();
    if ((int)threadIdx.x < V) {
        int v = threadIdx.x;
        float mx = -1e30f;
        for (int w = 0; w < V; ++w) mx = fmaxf(mx, S[v * V + w]);
        float ex[V]; float sum = 0.f;
        for (int w = 0; w < V; ++w) { ex[w] = __expf(S[v * V + w] - mx); sum += ex[w]; }
        float inv = 1.f / sum;
        float* dst = gout + (((long)b * T + t) * V + v) * V;
        for (int w = 0; w < V; ++w) dst[w] = ex[w] * inv;
    }
}

// xa[b,c,v,t] = sum_w g[b,t,v,w] * x[b,c,w,t]
__global__ void k_applyg(const float* __restrict__ g, const bf16* __restrict__ xin,
                         bf16* __restrict__ xout, int C) {
    long idx = (long)blockIdx.x * blockDim.x + threadIdx.x;
    long tot = (long)BB * C * VT;
    if (idx >= tot) return;
    int t = (int)(idx % T); long r = idx / T;
    int v = (int)(r % V); r /= V;
    int c = (int)(r % C); int b = (int)(r / C);
    const float* gr = g + (((long)b * T + t) * V + v) * V;
    const bf16* xr = xin + ((long)b * C + c) * (long)VT + t;
    float acc = 0.f;
#pragma unroll 5
    for (int w = 0; w < V; ++w) acc += gr[w] * bf2f(xr[(long)w * T]);
    xout[idx] = f2bf(acc);
}

// temporal conv, kernel 3, dilation dil, 'same' pad, 42 -> 42 channels
__global__ __launch_bounds__(256) void k_tconv(const bf16* __restrict__ xin,
                                               const float* __restrict__ Wt,
                                               const float* __restrict__ bias,
                                               bf16* __restrict__ out, int dil) {
    __shared__ float w[BC * 3];
    __shared__ float bsh;
    int b = blockIdx.x / BC, o = blockIdx.x % BC;
    for (int i = threadIdx.x; i < BC * 3; i += blockDim.x) w[i] = Wt[o * BC * 3 + i];
    if (threadIdx.x == 0) bsh = bias[o];
    __syncthreads();
    for (int idx = threadIdx.x; idx < VT; idx += blockDim.x) {
        int v = idx / T, t = idx % T;
        float acc = bsh;
        for (int i = 0; i < BC; ++i) {
            const bf16* row = xin + (((long)b * BC + i) * V + v) * T;
#pragma unroll
            for (int k = 0; k < 3; ++k) {
                int tt = t + (k - 1) * dil;
                if ((unsigned)tt < (unsigned)T) acc += w[i * 3 + k] * bf2f(row[tt]);
            }
        }
        out[(((long)b * BC + o) * V + v) * T + t] = f2bf(acc);
    }
}

__global__ void k_maxpool(const bf16* __restrict__ in, bf16* __restrict__ out) {
    long idx = (long)blockIdx.x * blockDim.x + threadIdx.x;
    long tot = (long)BB * BC * VT;
    if (idx >= tot) return;
    int t = (int)(idx % T);
    float m = bf2f(in[idx]);
    if (t > 0)     m = fmaxf(m, bf2f(in[idx - 1]));
    if (t < T - 1) m = fmaxf(m, bf2f(in[idx + 1]));
    out[idx] = f2bf(m);
}

// ---------------------------------------------------------------------------
// Host side
// ---------------------------------------------------------------------------
// Input leaf order: jax tree_flatten (sorted dict keys, recursive); x is last
// (also located by its unique size as a fallback).
enum {
    I_BNR_B = 0, I_BNR_G = 1,
    I_BR0 = 2,   // + j*8 : bn1_b, bn1_g, bn2_b, bn2_g, c1_W, c1_b, tc_W, tc_b
    I_G1_W = 34, I_G1_B = 35, I_G2_W = 36, I_G2_B = 37,
    I_GCN1_BN_B = 38, I_GCN1_BN_G = 39, I_GCN1_W = 40, I_GCN1_W1 = 41, I_GCN1_W1B = 42,
    I_GCN2_BN_B = 43, I_GCN2_BN_G = 44, I_GCN2_W = 45, I_GCN2_W1 = 46, I_GCN2_W1B = 47,
    I_GCN3_BN_B = 48, I_GCN3_BN_G = 49, I_GCN3_W = 50, I_GCN3_W1 = 51, I_GCN3_W1B = 52,
    I_MP_BN1_B = 53, I_MP_BN1_G = 54, I_MP_BN2_B = 55, I_MP_BN2_G = 56, I_MP_C1W = 57, I_MP_C1B = 58,
    I_MRES_W = 59, I_MRES_B = 60, I_MRES_BN_B = 61, I_MRES_BN_G = 62,
    I_P1_BN_B = 63, I_P1_BN_G = 64, I_P1_C1W = 65, I_P1_C1B = 66,
    I_RES_W = 67, I_RES_B = 68, I_UP_W = 69, I_UP_B = 70, I_X = 71
};

extern "C" void kernel_launch(void* const* d_in, const int* in_sizes, int n_in,
                              void* d_out, int out_size, void* d_ws, size_t ws_size,
                              hipStream_t stream) {
    auto P = [&](int i) { return (const float*)d_in[i]; };
    int xi = (n_in > I_X) ? I_X : n_in - 1;
    for (int i = 0; i < n_in; ++i)
        if (in_sizes[i] == BB * CIN * V * T) { xi = i; break; }

    // ---- workspace carve -------------------------------------------------
    char* base = (char*)d_ws;
    size_t off = 0;
    auto alloc = [&](size_t bytes) -> void* {
        void* p = base + off;
        off = (off + bytes + 255) & ~(size_t)255;
        return p;
    };
    bf16* XB  = (bf16*)alloc((size_t)BB * CIN * VT * 2);       // bf16(x)
    bf16* RES = (bf16*)alloc((size_t)BB * 256 * VT * 2);       // residual
    bf16* C1  = (bf16*)alloc((size_t)BB * 256 * VT * 2);       // g1 / t / mres
    bf16* D1  = (bf16*)alloc((size_t)BB * 256 * VT * 2);       // g2 / y
    float* G  = (float*)alloc((size_t)BB * T * V * V * 4);     // affinity
    bf16* FB  = (bf16*)alloc((size_t)BB * 256 * VT * 2);       // xa / concat
    bf16* IB  = (bf16*)alloc((size_t)BB * 256 * VT * 2);       // h1 / h3
    bf16* JB  = (bf16*)alloc((size_t)BB * 256 * VT * 2);       // h2 / h
    bf16* Y1  = (bf16*)alloc((size_t)BB * BC * VT * 2);
    bf16* Y2  = (bf16*)alloc((size_t)BB * BC * VT * 2);
    float* MEAN = (float*)alloc(6400 * 4);
    float* RSIG = (float*)alloc(6400 * 4);
    bf16* resWp  = (bf16*)alloc(256 * 128 * 2);
    bf16* g1Wp   = (bf16*)alloc(256 * 128 * 2);
    bf16* g2Wp   = (bf16*)alloc(256 * 128 * 2);
    bf16* gc1wP  = (bf16*)alloc(128 * 128 * 2);
    bf16* gc1w1P = (bf16*)alloc(128 * 128 * 2);
    bf16* gc2wP  = (bf16*)alloc(256 * 128 * 2);
    bf16* gc2w1P = (bf16*)alloc(256 * 128 * 2);
    bf16* gc3wP  = (bf16*)alloc(256 * 256 * 2);
    bf16* gc3w1P = (bf16*)alloc(256 * 256 * 2);
    bf16* brWp[4];
    for (int j = 0; j < 4; ++j) brWp[j] = (bf16*)alloc(48 * 256 * 2);
    bf16* mpWp   = (bf16*)alloc(48 * 256 * 2);
    bf16* p1Wp   = (bf16*)alloc(48 * 256 * 2);
    bf16* mresWp = (bf16*)alloc(256 * 256 * 2);
    bf16* upWp   = (bf16*)alloc(256 * 256 * 2);
    (void)ws_size; (void)out_size;

    // ---- helpers ---------------------------------------------------------
    auto cvt = [&](const float* W, bf16* Wp, int O, int K, int Opad, int Kpad) {
        int tot = Opad * Kpad;
        k_cvt_w<<<dim3((tot + 255) / 256), dim3(256), 0, stream>>>(W, Wp, O, K, Opad, Kpad);
    };
    auto gemmB = [&](const bf16* Wp, const bf16* X, const float* bias, const bf16* add,
                     bf16* out, int Kpad, int O, int Opad) {
        dim3 g((unsigned)(NTOT / 128), (unsigned)(Opad / 16)), blk(32);
        if (bias) k_gemm<bf16, true, false><<<g, blk, 0, stream>>>(Wp, X, bias, nullptr, out, Kpad, O);
        else      k_gemm<bf16, false, true><<<g, blk, 0, stream>>>(Wp, X, nullptr, add, out, Kpad, O);
    };
    auto stats = [&](const bf16* x, int CH, int L) {
        k_bn_stats<<<dim3(CH), dim3(256), 0, stream>>>(x, CH, L, MEAN, RSIG);
    };
    auto apply = [&](const bf16* x, int CH, int L, const float* gam, const float* bet,
                     bool relu, const bf16* add, bf16* out, int OCT, int OCO) {
        long tot = (long)BB * CH * L;
        dim3 g((unsigned)((tot + 255) / 256)), blk(256);
        if (relu && add) k_bn_apply<true, true><<<g, blk, 0, stream>>>(x, CH, L, MEAN, RSIG, gam, bet, add, out, OCT, OCO);
        else if (relu)   k_bn_apply<true, false><<<g, blk, 0, stream>>>(x, CH, L, MEAN, RSIG, gam, bet, nullptr, out, OCT, OCO);
        else if (add)    k_bn_apply<false, true><<<g, blk, 0, stream>>>(x, CH, L, MEAN, RSIG, gam, bet, add, out, OCT, OCO);
        else             k_bn_apply<false, false><<<g, blk, 0, stream>>>(x, CH, L, MEAN, RSIG, gam, bet, nullptr, out, OCT, OCO);
    };
    auto applyg = [&](const bf16* xin, bf16* xout, int C) {
        long tot = (long)BB * C * VT;
        k_applyg<<<dim3((unsigned)((tot + 255) / 256)), dim3(256), 0, stream>>>(G, xin, xout, C);
    };

    // ---- weight conversion (padded bf16) ---------------------------------
    cvt(P(I_RES_W), resWp, 256, 128, 256, 128);
    cvt(P(I_G1_W), g1Wp, 256, 128, 256, 128);
    cvt(P(I_G2_W), g2Wp, 256, 128, 256, 128);
    cvt(P(I_GCN1_W), gc1wP, 128, 128, 128, 128);
    cvt(P(I_GCN1_W1), gc1w1P, 128, 128, 128, 128);
    cvt(P(I_GCN2_W), gc2wP, 256, 128, 256, 128);
    cvt(P(I_GCN2_W1), gc2w1P, 256, 128, 256, 128);
    cvt(P(I_GCN3_W), gc3wP, 256, 256, 256, 256);
    cvt(P(I_GCN3_W1), gc3w1P, 256, 256, 256, 256);
    for (int j = 0; j < 4; ++j) cvt(P(I_BR0 + j * 8 + 4), brWp[j], BC, 256, 48, 256);
    cvt(P(I_MP_C1W), mpWp, BC, 256, 48, 256);
    cvt(P(I_P1_C1W), p1Wp, BC, 256, 48, 256);
    cvt(P(I_MRES_W), mresWp, 252, 256, 256, 256);
    cvt(P(I_UP_W), upWp, 256, 252, 256, 256);

    // ---- front end -------------------------------------------------------
    {
        long n = (long)BB * CIN * VT;
        k_f2b<<<dim3((unsigned)((n + 255) / 256)), dim3(256), 0, stream>>>(P(xi), XB, n);
    }
    gemmB(resWp, XB, P(I_RES_B), nullptr, RES, 128, 256, 256);    // residual
    gemmB(g1Wp, XB, P(I_G1_B), nullptr, C1, 128, 256, 256);       // g1
    gemmB(g2Wp, XB, P(I_G2_B), nullptr, D1, 128, 256, 256);       // g2
    k_gmat<<<dim3(BB * T), dim3(128), 0, stream>>>(C1, D1, G);    // softmax affinity

    // ---- GCN stack -------------------------------------------------------
    // gcn1: 128 -> 128
    applyg(XB, FB, 128);
    gemmB(gc1w1P, XB, P(I_GCN1_W1B), nullptr, C1, 128, 128, 128); // t = w1*x + b
    gemmB(gc1wP, FB, nullptr, C1, D1, 128, 128, 128);             // y = w*xa + t
    stats(D1, 128, VT);
    apply(D1, 128, VT, P(I_GCN1_BN_G), P(I_GCN1_BN_B), true, nullptr, IB, 128, 0);
    // gcn2: 128 -> 256
    applyg(IB, FB, 128);
    gemmB(gc2w1P, IB, P(I_GCN2_W1B), nullptr, C1, 128, 256, 256);
    gemmB(gc2wP, FB, nullptr, C1, D1, 128, 256, 256);
    stats(D1, 256, VT);
    apply(D1, 256, VT, P(I_GCN2_BN_G), P(I_GCN2_BN_B), true, nullptr, JB, 256, 0);
    // gcn3: 256 -> 256
    applyg(JB, FB, 256);
    gemmB(gc3w1P, JB, P(I_GCN3_W1B), nullptr, C1, 256, 256, 256);
    gemmB(gc3wP, FB, nullptr, C1, D1, 256, 256, 256);
    stats(D1, 256, VT);
    apply(D1, 256, VT, P(I_GCN3_BN_G), P(I_GCN3_BN_B), true, nullptr, IB, 256, 0);

    // ---- norm_act (BatchNorm1d over C*V channels) + relu + residual ------
    stats(IB, 6400, T);
    apply(IB, 6400, T, P(I_BNR_G), P(I_BNR_B), true, RES, JB, 6400, 0);   // JB = h

    // ---- multi-scale temporal branches into padded concat (FB, 256 ch) ---
    {
        long n = (long)BB * 4 * VT;
        k_zeropad<<<dim3((unsigned)((n + 255) / 256)), dim3(256), 0, stream>>>(FB);
    }
    for (int j = 0; j < 4; ++j) {
        int dil = j + 1, offc = j * BC;
        gemmB(brWp[j], JB, P(I_BR0 + j * 8 + 5), nullptr, Y1, 256, BC, 48);
        stats(Y1, BC, VT);
        apply(Y1, BC, VT, P(I_BR0 + j * 8 + 1), P(I_BR0 + j * 8 + 0), true, nullptr, Y2, BC, 0);
        k_tconv<<<dim3(BB * BC), dim3(256), 0, stream>>>(Y2, P(I_BR0 + j * 8 + 6), P(I_BR0 + j * 8 + 7), Y1, dil);
        stats(Y1, BC, VT);
        apply(Y1, BC, VT, P(I_BR0 + j * 8 + 3), P(I_BR0 + j * 8 + 2), false, nullptr, FB, 256, offc);
    }
    // maxpool branch
    gemmB(mpWp, JB, P(I_MP_C1B), nullptr, Y1, 256, BC, 48);
    stats(Y1, BC, VT);
    apply(Y1, BC, VT, P(I_MP_BN1_G), P(I_MP_BN1_B), true, nullptr, Y2, BC, 0);
    {
        long n = (long)BB * BC * VT;
        k_maxpool<<<dim3((unsigned)((n + 255) / 256)), dim3(256), 0, stream>>>(Y2, Y1);
    }
    stats(Y1, BC, VT);
    apply(Y1, BC, VT, P(I_MP_BN2_G), P(I_MP_BN2_B), false, nullptr, FB, 256, 4 * BC);
    // 1x1 branch
    gemmB(p1Wp, JB, P(I_P1_C1B), nullptr, Y1, 256, BC, 48);
    stats(Y1, BC, VT);
    apply(Y1, BC, VT, P(I_P1_BN_G), P(I_P1_BN_B), false, nullptr, FB, 256, 5 * BC);
    // mres: bn(conv1x1(h)) added into concat in place
    gemmB(mresWp, JB, P(I_MRES_B), nullptr, C1, 256, 252, 256);
    stats(C1, 252, VT);
    apply(C1, 252, VT, P(I_MRES_BN_G), P(I_MRES_BN_B), false, FB, FB, 256, 0);

    // ---- final up-projection + residual, fp32 out ------------------------
    {
        dim3 g((unsigned)(NTOT / 128), 16), blk(32);
        k_gemm<float, true, true><<<g, blk, 0, stream>>>(upWp, FB, P(I_UP_B), RES,
                                                         (float*)d_out, 256, 256);
    }
}
